// QuantumAttention_6725918785913
// MI455X (gfx1250) — matrix operand order repaired and checked by reading, not compiled, not run on hardware
//
#include <hip/hip_runtime.h>
#include <math.h>

typedef __attribute__((ext_vector_type(16))) _Float16 v16h;
typedef __attribute__((ext_vector_type(8)))  float    v8f;

#define NQ 4
#define IN_DIM 96

// Fast hardware sin/cos (v_sin_f32 / v_cos_f32). All angles in this workload
// are O(pi), well inside the accurate range of the native ops.
__device__ __forceinline__ float fcos(float x) { return __cosf(x); }
__device__ __forceinline__ float fsin(float x) { return __sinf(x); }

// =====================================================================
// Kernel A: xq = x_text @ W_text^T + b_text ; xk = x_image @ W_image^T + b_image
// One wave (32 threads) per 16-row tile; N padded 4 -> 16; K = 96 = 3 x 32.
// Uses v_wmma_f32_16x16x32_f16 (f32 accumulate).
// =====================================================================
__global__ __launch_bounds__(32)
void proj_wmma_kernel(const float* __restrict__ x_text,
                      const float* __restrict__ x_image,
                      const float* __restrict__ W_text,
                      const float* __restrict__ b_text,
                      const float* __restrict__ W_image,
                      const float* __restrict__ b_image,
                      float* __restrict__ xq,
                      float* __restrict__ xk,
                      int B)
{
  const int lane = threadIdx.x & 31;
  const int tile = blockIdx.x;
  const int mcol = lane & 15;   // A-matrix row within tile; C/D column
  const int hi   = lane >> 4;   // lane group 0/1

  const int arow  = tile * 16 + mcol;
  const int arowc = arow < B ? arow : (B - 1);

  // Unconditional (clamped) W row + multiplicative mask: avoids per-element
  // exec-mask branches around the padded-N loads.
  const int   wrow  = (mcol < NQ) ? mcol : 0;
  const float wmask = (mcol < NQ) ? 1.f : 0.f;

  v8f accQ = {};
  v8f accK = {};

  #pragma unroll
  for (int k0 = 0; k0 < IN_DIM; k0 += 32) {
    v16h aQ, aK, bQ, bK;
    // A fragment: 16x32 f16; lane m = lane%16; K per ISA 16-bit A layout
    #pragma unroll
    for (int j = 0; j < 16; ++j) {
      int kk = k0 + (j >> 3) * 16 + hi * 8 + ((j >> 1) & 3) * 2 + (j & 1);
      aQ[j] = (_Float16)x_text [arowc * IN_DIM + kk];
      aK[j] = (_Float16)x_image[arowc * IN_DIM + kk];
    }
    // B fragment: 32x16 f16, B[K][N] = W[N][K]; N = lane%16, K = (j/8)*16 + (lane/16)*8 + j%8
    #pragma unroll
    for (int j = 0; j < 16; ++j) {
      int kk = k0 + (j >> 3) * 16 + hi * 8 + (j & 7);
      bQ[j] = (_Float16)(wmask * W_text [wrow * IN_DIM + kk]);
      bK[j] = (_Float16)(wmask * W_image[wrow * IN_DIM + kk]);
    }
    accQ = __builtin_amdgcn_wmma_f32_16x16x32_f16(false, aQ, false, bQ,
                                                  (short)0, accQ, false, false);
    accK = __builtin_amdgcn_wmma_f32_16x16x32_f16(false, aK, false, bK,
                                                  (short)0, accK, false, false);
  }

  // C/D layout: VGPR r, lane L: N = L%16, M = (L/16)*8 + r
  #pragma unroll
  for (int r = 0; r < 8; ++r) {
    int grow = tile * 16 + hi * 8 + r;
    if (mcol < NQ && grow < B) {
      xq[grow * NQ + mcol] = accQ[r] + b_text[mcol];
      xk[grow * NQ + mcol] = accK[r] + b_image[mcol];
    }
  }
}

// =====================================================================
// 8-qubit statevector distributed over one wave32:
// flat index i = lane*8 + j ; bits[2:0] local (j), bits[7:3] = lane.
// Qubit w <-> bit position p = 7 - w (C-order flattening of (2,)*8 tensor).
// =====================================================================
__device__ __forceinline__ float wsum(float v) {
  #pragma unroll
  for (int off = 16; off > 0; off >>= 1) v += __shfl_xor(v, off, 32);
  return v;
}

// fr/fi[j] = old amplitude at global index (lane*8+j) ^ f
__device__ __forceinline__ void fflip(const float (&sr)[8], const float (&si)[8],
                                      int f, float (&fr)[8], float (&fi)[8]) {
  const int flo = f & 7, fhi = f >> 3;
  #pragma unroll
  for (int j = 0; j < 8; ++j) {
    float r = sr[j ^ flo];
    float i = si[j ^ flo];
    if (fhi) { r = __shfl_xor(r, fhi, 32); i = __shfl_xor(i, fhi, 32); }
    fr[j] = r; fi[j] = i;
  }
}

// RZ(t) = diag(e^{-i t/2}, e^{+i t/2})
__device__ __forceinline__ void rz8(float (&sr)[8], float (&si)[8],
                                    int lane, int p, float t) {
  float h = 0.5f * t, c = fcos(h), s = fsin(h);
  #pragma unroll
  for (int j = 0; j < 8; ++j) {
    int idx = (lane << 3) | j;
    float ei = ((idx >> p) & 1) ? s : -s;   // phase imag part
    float r = sr[j], im = si[j];
    sr[j] = r * c - im * ei;
    si[j] = r * ei + im * c;
  }
}

// new = c*cur - i*s*flip  (RX is symmetric in the pair)
__device__ __forceinline__ void rx8(float (&sr)[8], float (&si)[8],
                                    int lane, int p, float t) {
  (void)lane;
  float h = 0.5f * t, c = fcos(h), s = fsin(h);
  float fr[8], fi[8];
  fflip(sr, si, 1 << p, fr, fi);
  #pragma unroll
  for (int j = 0; j < 8; ++j) {
    float r = sr[j], im = si[j];
    sr[j] = c * r  + s * fi[j];
    si[j] = c * im - s * fr[j];
  }
}

__device__ __forceinline__ void crx8(float (&sr)[8], float (&si)[8],
                                     int lane, int pc, int pt, float t) {
  float h = 0.5f * t, c = fcos(h), s = fsin(h);
  float fr[8], fi[8];
  fflip(sr, si, 1 << pt, fr, fi);     // shuffles outside any divergence
  #pragma unroll
  for (int j = 0; j < 8; ++j) {
    int idx = (lane << 3) | j;
    bool cb = ((idx >> pc) & 1) != 0;
    float r = sr[j], im = si[j];
    float nr = c * r  + s * fi[j];
    float ni = c * im - s * fr[j];
    sr[j] = cb ? nr : r;
    si[j] = cb ? ni : im;
  }
}

// IsingXX: new(x) = c*old(x) - i*s*old(x ^ (bit1|bit2)) for all x
__device__ __forceinline__ void ixx8(float (&sr)[8], float (&si)[8],
                                     int lane, int p1, int p2, float t) {
  (void)lane;
  float h = 0.5f * t, c = fcos(h), s = fsin(h);
  float fr[8], fi[8];
  fflip(sr, si, (1 << p1) | (1 << p2), fr, fi);
  #pragma unroll
  for (int j = 0; j < 8; ++j) {
    float r = sr[j], im = si[j];
    sr[j] = c * r  + s * fi[j];
    si[j] = c * im - s * fr[j];
  }
}

__device__ __forceinline__ void cnot8(float (&sr)[8], float (&si)[8],
                                      int lane, int pc, int pt) {
  float fr[8], fi[8];
  fflip(sr, si, 1 << pt, fr, fi);
  #pragma unroll
  for (int j = 0; j < 8; ++j) {
    int idx = (lane << 3) | j;
    bool cb = ((idx >> pc) & 1) != 0;
    sr[j] = cb ? fr[j] : sr[j];
    si[j] = cb ? fi[j] : si[j];
  }
}

__device__ __forceinline__ float expz8(const float (&sr)[8], const float (&si)[8],
                                       int lane, int p) {
  float acc = 0.f;
  #pragma unroll
  for (int j = 0; j < 8; ++j) {
    int idx = (lane << 3) | j;
    float pr = sr[j] * sr[j] + si[j] * si[j];
    acc += ((idx >> p) & 1) ? -pr : pr;
  }
  return wsum(acc);
}

__device__ __forceinline__ float expx8(const float (&sr)[8], const float (&si)[8],
                                       int lane, int p) {
  float fr[8], fi[8];
  fflip(sr, si, 1 << p, fr, fi);
  float acc = 0.f;
  #pragma unroll
  for (int j = 0; j < 8; ++j)
    acc += sr[j] * fr[j] + si[j] * fi[j];   // Re(conj(cur)*flip)
  return wsum(acc);
}

__device__ __forceinline__ void rot_layer8(float (&sr)[8], float (&si)[8], int lane,
                                           const float* __restrict__ rot,
                                           int offset, int start) {
  #pragma unroll
  for (int i = 0; i < NQ; ++i) {
    int p = 7 - (offset + i);
    rz8(sr, si, lane, p, rot[start + 3 * i]);
    rx8(sr, si, lane, p, rot[start + 3 * i + 1]);
    rz8(sr, si, lane, p, rot[start + 3 * i + 2]);
  }
}

__device__ __forceinline__ void two_ring8(float (&sr)[8], float (&si)[8], int lane,
                                          const float* __restrict__ rot,
                                          const float* __restrict__ crx, int offset) {
  rot_layer8(sr, si, lane, rot, offset, 0);
  #pragma unroll
  for (int i = 0; i < NQ; ++i)
    crx8(sr, si, lane, 7 - (offset + i), 7 - (offset + ((i + 1) & 3)), crx[i]);
  #pragma unroll
  for (int i = 0; i < NQ; ++i)
    ixx8(sr, si, lane, 7 - (offset + i), 7 - (offset + ((i + 1) & 3)), crx[NQ + i]);
  rot_layer8(sr, si, lane, rot, offset, 12);
}

// One batch element per wave. amp[b][i] = sqrt(z_i^2 + x_i^2).
__global__ __launch_bounds__(256)
void score_kernel(const float* __restrict__ xq, const float* __restrict__ xk,
                  const float* __restrict__ wq_rot, const float* __restrict__ wq_crx,
                  const float* __restrict__ wk_rot, const float* __restrict__ wk_crx,
                  const float* __restrict__ wc_rot, const float* __restrict__ wc_crx,
                  const float* __restrict__ wc_crx2,
                  float* __restrict__ amp, int B)
{
  const int lane = threadIdx.x & 31;
  const int b = blockIdx.x * (blockDim.x >> 5) + (threadIdx.x >> 5);
  if (b >= B) return;   // uniform per wave

  float sr[8], si[8];
  #pragma unroll
  for (int j = 0; j < 8; ++j) { sr[j] = (lane == 0 && j == 0) ? 1.f : 0.f; si[j] = 0.f; }

  #pragma unroll
  for (int i = 0; i < NQ; ++i) rx8(sr, si, lane, 7 - i, xq[b * NQ + i]);
  two_ring8(sr, si, lane, wq_rot, wq_crx, 0);

  #pragma unroll
  for (int i = 0; i < NQ; ++i) rx8(sr, si, lane, 7 - (NQ + i), xk[b * NQ + i]);
  two_ring8(sr, si, lane, wk_rot, wk_crx, NQ);

  #pragma unroll
  for (int i = 0; i < NQ; ++i) {
    crx8(sr, si, lane, 7 - i, 3 - i, wc_crx[i]);       // CRX(i -> i+4)
    crx8(sr, si, lane, 3 - i, 7 - i, wc_crx[NQ + i]);  // CRX(i+4 -> i)
  }
  #pragma unroll
  for (int i = 0; i < NQ; ++i)
    crx8(sr, si, lane, 7 - i, 3 - ((i + 1) & 3), wc_crx2[i]);       // i -> (i+1)%4+4
  #pragma unroll
  for (int i = 0; i < NQ; ++i)
    crx8(sr, si, lane, 3 - i, 7 - ((i + 1) & 3), wc_crx2[NQ + i]);  // i+4 -> (i+1)%4
  #pragma unroll
  for (int i = 0; i < NQ; ++i) {
    cnot8(sr, si, lane, 7 - i, 3 - i);
    cnot8(sr, si, lane, 3 - i, 7 - i);
  }
  rot_layer8(sr, si, lane, wc_rot, 0, 0);

  #pragma unroll
  for (int i = 0; i < NQ; ++i) {
    int p = 7 - i;
    float z = expz8(sr, si, lane, p);
    float x = expx8(sr, si, lane, p);
    float a = __builtin_sqrtf(z * z + x * x);
    if (lane == 0) amp[b * NQ + i] = a;
  }
}

// =====================================================================
// Kernel C: 4-qubit value circuit, one batch element per thread.
// Qubit w <-> bit p = 3 - w.
// =====================================================================
__device__ __forceinline__ void rxpair(float (&sr)[16], float (&si)[16],
                                       int i0, int i1, float c, float s) {
  float ar = sr[i0], ai = si[i0], br = sr[i1], bi = si[i1];
  sr[i0] = c * ar + s * bi;  si[i0] = c * ai - s * br;
  sr[i1] = c * br + s * ai;  si[i1] = c * bi - s * ar;
}

__device__ __forceinline__ void rz16(float (&sr)[16], float (&si)[16], int p, float t) {
  float h = 0.5f * t, c = fcos(h), s = fsin(h);
  #pragma unroll
  for (int idx = 0; idx < 16; ++idx) {
    float ei = ((idx >> p) & 1) ? s : -s;
    float r = sr[idx], im = si[idx];
    sr[idx] = r * c - im * ei;
    si[idx] = r * ei + im * c;
  }
}

__device__ __forceinline__ void rx16(float (&sr)[16], float (&si)[16], int p, float t) {
  float h = 0.5f * t, c = fcos(h), s = fsin(h);
  int m = 1 << p;
  #pragma unroll
  for (int idx = 0; idx < 16; ++idx)
    if (!(idx & m)) rxpair(sr, si, idx, idx | m, c, s);
}

__device__ __forceinline__ void crx16(float (&sr)[16], float (&si)[16],
                                      int pc, int pt, float t) {
  float h = 0.5f * t, c = fcos(h), s = fsin(h);
  int mc = 1 << pc, mt = 1 << pt;
  #pragma unroll
  for (int idx = 0; idx < 16; ++idx)
    if ((idx & mc) && !(idx & mt)) rxpair(sr, si, idx, idx | mt, c, s);
}

__device__ __forceinline__ void ixx16(float (&sr)[16], float (&si)[16],
                                      int p1, int p2, float t) {
  float h = 0.5f * t, c = fcos(h), s = fsin(h);
  int m = (1 << p1) | (1 << p2);
  #pragma unroll
  for (int idx = 0; idx < 16; ++idx)
    if (idx < (idx ^ m)) rxpair(sr, si, idx, idx ^ m, c, s);
}

__device__ __forceinline__ void cnot16(float (&sr)[16], float (&si)[16],
                                       int pc, int pt) {
  int mc = 1 << pc, mt = 1 << pt;
  #pragma unroll
  for (int idx = 0; idx < 16; ++idx)
    if ((idx & mc) && !(idx & mt)) {
      int i1 = idx | mt;
      float tr = sr[idx], ti = si[idx];
      sr[idx] = sr[i1]; si[idx] = si[i1];
      sr[i1] = tr;      si[i1] = ti;
    }
}

__global__ __launch_bounds__(256)
void value_kernel(const float* __restrict__ xk, const float* __restrict__ amp,
                  const float* __restrict__ wv_rot, const float* __restrict__ wv_crx,
                  const float* __restrict__ gates,
                  float* __restrict__ out, int B)
{
  int b = blockIdx.x * blockDim.x + threadIdx.x;
  if (b >= B) return;

  float sr[16], si[16];
  #pragma unroll
  for (int i = 0; i < 16; ++i) { sr[i] = (i == 0) ? 1.f : 0.f; si[i] = 0.f; }

  #pragma unroll
  for (int i = 0; i < NQ; ++i) rx16(sr, si, 3 - i, xk[b * NQ + i]);

  // two_ring(wv_rot, wv_crx)
  #pragma unroll
  for (int i = 0; i < NQ; ++i) {
    int p = 3 - i;
    rz16(sr, si, p, wv_rot[3 * i]);
    rx16(sr, si, p, wv_rot[3 * i + 1]);
    rz16(sr, si, p, wv_rot[3 * i + 2]);
  }
  #pragma unroll
  for (int i = 0; i < NQ; ++i)
    crx16(sr, si, 3 - i, 3 - ((i + 1) & 3), wv_crx[i]);
  #pragma unroll
  for (int i = 0; i < NQ; ++i)
    ixx16(sr, si, 3 - i, 3 - ((i + 1) & 3), wv_crx[NQ + i]);
  #pragma unroll
  for (int i = 0; i < NQ; ++i) {
    int p = 3 - i;
    rz16(sr, si, p, wv_rot[12 + 3 * i]);
    rx16(sr, si, p, wv_rot[12 + 3 * i + 1]);
    rz16(sr, si, p, wv_rot[12 + 3 * i + 2]);
  }

  #pragma unroll
  for (int i = 0; i < NQ; ++i)
    rx16(sr, si, 3 - i, tanhf(amp[b * NQ + i]) * gates[i]);
  #pragma unroll
  for (int i = 0; i < NQ; ++i)
    cnot16(sr, si, 3 - i, 3 - ((i + 1) & 3));

  #pragma unroll
  for (int w = 0; w < NQ; ++w) {
    int p = 3 - w, m = 1 << p;
    float z = 0.f, x = 0.f;
    #pragma unroll
    for (int idx = 0; idx < 16; ++idx) {
      float pr = sr[idx] * sr[idx] + si[idx] * si[idx];
      z += ((idx >> p) & 1) ? -pr : pr;
      int f = idx ^ m;
      x += sr[idx] * sr[f] + si[idx] * si[f];
    }
    out[b * 8 + w]      = z;
    out[b * 8 + NQ + w] = x;
  }
}

// =====================================================================
extern "C" void kernel_launch(void* const* d_in, const int* in_sizes, int n_in,
                              void* d_out, int out_size, void* d_ws, size_t ws_size,
                              hipStream_t stream)
{
  (void)n_in; (void)out_size; (void)ws_size;

  const float* x_text  = (const float*)d_in[0];
  const float* x_image = (const float*)d_in[1];
  const float* W_text  = (const float*)d_in[2];
  const float* b_text  = (const float*)d_in[3];
  const float* W_image = (const float*)d_in[4];
  const float* b_image = (const float*)d_in[5];
  const float* wq_rot  = (const float*)d_in[6];
  const float* wq_crx  = (const float*)d_in[7];
  const float* wk_rot  = (const float*)d_in[8];
  const float* wk_crx  = (const float*)d_in[9];
  const float* wv_rot  = (const float*)d_in[10];
  const float* wv_crx  = (const float*)d_in[11];
  const float* wc_rot  = (const float*)d_in[12];
  const float* wc_crx  = (const float*)d_in[13];
  const float* wc_crx2 = (const float*)d_in[14];
  const float* gates   = (const float*)d_in[15];
  float* out = (float*)d_out;

  const int B = in_sizes[0] / IN_DIM;

  float* xq  = (float*)d_ws;
  float* xk  = xq + (size_t)B * NQ;
  float* amp = xk + (size_t)B * NQ;

  // Projections via WMMA: one wave per 16-row tile.
  int tiles = (B + 15) / 16;
  proj_wmma_kernel<<<tiles, 32, 0, stream>>>(x_text, x_image, W_text, b_text,
                                             W_image, b_image, xq, xk, B);

  // Score circuit: one batch element per wave32 (8 waves / block).
  const int wavesPerBlock = 8;
  int sblocks = (B + wavesPerBlock - 1) / wavesPerBlock;
  score_kernel<<<sblocks, wavesPerBlock * 32, 0, stream>>>(
      xq, xk, wq_rot, wq_crx, wk_rot, wk_crx, wc_rot, wc_crx, wc_crx2, amp, B);

  // Value circuit: one batch element per thread.
  value_kernel<<<(B + 255) / 256, 256, 0, stream>>>(xk, amp, wv_rot, wv_crx,
                                                    gates, out, B);
}
